// MultiHeadAttention_79860621902536
// MI455X (gfx1250) — matrix operand order repaired and checked by reading, not compiled, not run on hardware
//
#include <hip/hip_runtime.h>
#include <hip/hip_bf16.h>

// ---------------------------------------------------------------------------
// MHA forward for MI455X (gfx1250, wave32, WMMA f16 -> f32 accumulate)
//   B=4, S=2048, D=1024, H=16, HD=64
// Pipeline: fp32->f16 convert, QKV projection GEMMs (WMMA), flash-attention
// (WMMA + online softmax), output projection GEMM (WMMA, f32 out).
// ---------------------------------------------------------------------------

typedef _Float16 v16h __attribute__((ext_vector_type(16)));
typedef float    v8f  __attribute__((ext_vector_type(8)));

#define Bsz 4
#define Sq  2048
#define Dm  1024
#define Hn  16
#define HDm 64
#define Mrows (Bsz * Sq)   // 8192

__device__ __forceinline__ v8f wmma16(v16h a, v16h b, v8f c) {
    // v_wmma_f32_16x16x32_f16 : D = A(16x32 f16) x B(32x16 f16) + C(16x16 f32)
    return __builtin_amdgcn_wmma_f32_16x16x32_f16(
        false, a, false, b, (short)0, c, false, false);
}

// A-fragment (16x32 f16, MxK). Lane L: m = L&15, half = L>>4.
// v16h e=0..7  -> K = half*8 + e        (contiguous 16B)
//      e=8..15 -> K = 16 + half*8 + e-8 (contiguous 16B)
__device__ __forceinline__ v16h load_a_frag(const _Float16* base, int ldk,
                                            int m0, int k0, int lane) {
    int m    = m0 + (lane & 15);
    int half = lane >> 4;
    const _Float16* p = base + (size_t)m * ldk + k0 + half * 8;
    union { v16h v; uint4 u[2]; } f;
    f.u[0] = *reinterpret_cast<const uint4*>(p);
    f.u[1] = *reinterpret_cast<const uint4*>(p + 16);
    return f.v;
}

// B-fragment (32x16 f16, KxN), operand stored row-major [N][K] (K contiguous).
// Lane L: n = L&15, e=0..7 -> K = (L>>4)*8 + e, e=8..15 -> K = 16 + (L>>4)*8 + e-8
__device__ __forceinline__ v16h load_b_frag(const _Float16* base, int ldk,
                                            int n0, int k0, int lane) {
    int n  = n0 + (lane & 15);
    int kh = lane >> 4;
    const _Float16* p = base + (size_t)n * ldk + k0 + kh * 8;
    union { v16h v; uint4 u[2]; } f;
    f.u[0] = *reinterpret_cast<const uint4*>(p);
    f.u[1] = *reinterpret_cast<const uint4*>(p + 16);
    return f.v;
}

// ---------------------------------------------------------------------------
// fp32 -> f16 conversion
// ---------------------------------------------------------------------------
__global__ __launch_bounds__(256) void cvt_f32_f16(const float* __restrict__ src,
                                                   _Float16* __restrict__ dst,
                                                   int n) {
    int i = blockIdx.x * 256 + threadIdx.x;
    if (i < n) dst[i] = (_Float16)src[i];
}

// ---------------------------------------------------------------------------
// GEMM: Y[m,n] = sum_k X[m,k] * W[n,k] + bias[n]
//   X: [Mrows, Dm] f16 row-major,  W: [Dm, Dm] f16 row-major (K contiguous)
//   mode 0: outH laid out [B,H,S,HD]    (Q, K projections)
//   mode 1: outH laid out [B,H,HD,S]    (V projection, transposed for PV)
//   mode 2: outF laid out [B,S,D] fp32  (final output projection)
// Block = 128 threads = 4 waves in 2x2, each wave owns a 64x64 tile
// (16 accumulators -> 16 WMMAs per 32-wide k step).
// ---------------------------------------------------------------------------
__global__ __launch_bounds__(128) void proj_gemm(const _Float16* __restrict__ X,
                                                 const _Float16* __restrict__ W,
                                                 const float* __restrict__ bias,
                                                 _Float16* __restrict__ outH,
                                                 float* __restrict__ outF,
                                                 int mode) {
    const int lane = threadIdx.x & 31;
    const int wave = threadIdx.x >> 5;              // 0..3
    const int m0 = blockIdx.x * 128 + (wave >> 1) * 64;
    const int n0 = blockIdx.y * 128 + (wave & 1) * 64;

    v8f acc[4][4] = {};

    for (int k0 = 0; k0 < Dm; k0 += 32) {
        if (k0 + 32 < Dm) {   // prefetch next k-slab -> global_prefetch_b8
            __builtin_prefetch(&X[(size_t)(m0 + (lane & 31)) * Dm + k0 + 32], 0, 0);
            __builtin_prefetch(&W[(size_t)(n0 + (lane & 31)) * Dm + k0 + 32], 0, 0);
        }
        v16h a[4], b[4];
#pragma unroll
        for (int i = 0; i < 4; ++i) a[i] = load_a_frag(X, Dm, m0 + i * 16, k0, lane);
#pragma unroll
        for (int j = 0; j < 4; ++j) b[j] = load_b_frag(W, Dm, n0 + j * 16, k0, lane);
#pragma unroll
        for (int i = 0; i < 4; ++i)
#pragma unroll
            for (int j = 0; j < 4; ++j)
                acc[i][j] = wmma16(a[i], b[j], acc[i][j]);
    }

    // Epilogue. C/D layout: lane L, vgpr r -> M = r + (L>>4)*8, N = L&15.
    const int colL  = lane & 15;
    const int rowHB = (lane >> 4) * 8;
#pragma unroll
    for (int j = 0; j < 4; ++j) {
        const int n  = n0 + j * 16 + colL;
        const float bv = bias[n];
        const int h  = n >> 6;          // n / 64
        const int hd = n & 63;
#pragma unroll
        for (int i = 0; i < 4; ++i) {
#pragma unroll
            for (int r = 0; r < 8; ++r) {
                const int m = m0 + i * 16 + r + rowHB;
                const int b = m >> 11;          // m / 2048
                const int s = m & 2047;
                const float val = acc[i][j][r] + bv;
                if (mode == 0) {
                    outH[(((size_t)(b * Hn + h)) * Sq + s) * HDm + hd] = (_Float16)val;
                } else if (mode == 1) {
                    outH[(((size_t)(b * Hn + h)) * HDm + hd) * Sq + s] = (_Float16)val;
                } else {
                    outF[(size_t)m * Dm + n] = val;
                }
            }
        }
    }
}

// ---------------------------------------------------------------------------
// Flash attention. Q,K: [B,H,S,HD] f16;  Vt: [B,H,HD,S] f16;
// ctx out: [B,S,D] f16 (ready to be the A operand of the output projection).
// Block = 256 threads = 8 waves; each wave owns 16 query rows; block covers
// 128 rows of one (b,h). Keys streamed in chunks of 32:
//   scores (2x v8f) via 4 WMMAs, online softmax, P -> LDS -> A-fragment,
//   PV via 4 WMMAs into 16x64 f32 accumulators.
// ---------------------------------------------------------------------------
__global__ __launch_bounds__(256) void attn_kernel(const _Float16* __restrict__ Qh,
                                                   const _Float16* __restrict__ Kh,
                                                   const _Float16* __restrict__ Vt,
                                                   _Float16* __restrict__ ctx) {
    __shared__ _Float16 pbuf[8][16][40];   // per-wave 16x32 P tile, padded rows

    const int lane = threadIdx.x & 31;
    const int wave = threadIdx.x >> 5;
    const int bh = blockIdx.y;                       // 0..B*H-1
    const int q0 = blockIdx.x * 128 + wave * 16;

    const _Float16* qp = Qh + ((size_t)bh * Sq + q0) * HDm;
    const _Float16* kp = Kh + (size_t)bh * Sq * HDm;
    const _Float16* vp = Vt + (size_t)bh * HDm * Sq;

    // Q A-fragments: contraction over HD=64 -> two 16x32 chunks, loaded once.
    v16h qa0 = load_a_frag(qp, HDm, 0, 0, lane);
    v16h qa1 = load_a_frag(qp, HDm, 0, 32, lane);

    v8f  o[4] = {};
    float mrow[8], lrow[8];
#pragma unroll
    for (int r = 0; r < 8; ++r) { mrow[r] = -1e30f; lrow[r] = 0.0f; }

    const float scale = 0.125f;                      // 1/sqrt(64)
    const int mIdx = lane & 15;
    const int half = lane >> 4;

    for (int kc = 0; kc < Sq; kc += 32) {
        // ---- scores: S[16 x 32] = Q . K^T  (keys kc..kc+31) ----
        v16h b0a = load_b_frag(kp, HDm, kc,      0,  lane);
        v16h b0b = load_b_frag(kp, HDm, kc,      32, lane);
        v16h b1a = load_b_frag(kp, HDm, kc + 16, 0,  lane);
        v16h b1b = load_b_frag(kp, HDm, kc + 16, 32, lane);
        v8f s0 = {}, s1 = {};
        s0 = wmma16(qa0, b0a, s0);  s0 = wmma16(qa1, b0b, s0);
        s1 = wmma16(qa0, b1a, s1);  s1 = wmma16(qa1, b1b, s1);

        // ---- online softmax (row = r + half*8, spread over 16 lanes) ----
#pragma unroll
        for (int r = 0; r < 8; ++r) {
            float x0 = s0[r] * scale;
            float x1 = s1[r] * scale;
            float mx = fmaxf(x0, x1);
            mx = fmaxf(mx, __shfl_xor(mx, 1, 32));
            mx = fmaxf(mx, __shfl_xor(mx, 2, 32));
            mx = fmaxf(mx, __shfl_xor(mx, 4, 32));
            mx = fmaxf(mx, __shfl_xor(mx, 8, 32));
            float mnew = fmaxf(mrow[r], mx);
            float corr = __expf(mrow[r] - mnew);
            float p0 = __expf(x0 - mnew);
            float p1 = __expf(x1 - mnew);
            float ps = p0 + p1;
            ps += __shfl_xor(ps, 1, 32);
            ps += __shfl_xor(ps, 2, 32);
            ps += __shfl_xor(ps, 4, 32);
            ps += __shfl_xor(ps, 8, 32);
            lrow[r] = lrow[r] * corr + ps;
            mrow[r] = mnew;
#pragma unroll
            for (int t = 0; t < 4; ++t) o[t][r] *= corr;

            const int pr = r + half * 8;
            pbuf[wave][pr][mIdx]      = (_Float16)p0;
            pbuf[wave][pr][16 + mIdx] = (_Float16)p1;
        }

        asm volatile("s_wait_dscnt 0" ::: "memory");   // P tile visible in LDS

        // ---- P as A-fragment (16x32) from LDS ----
        union { v16h v; uint4 u[2]; } pa;
        pa.u[0] = *reinterpret_cast<const uint4*>(&pbuf[wave][mIdx][half * 8]);
        pa.u[1] = *reinterpret_cast<const uint4*>(&pbuf[wave][mIdx][16 + half * 8]);

        // ---- O += P . V   (V^T rows contiguous over keys) ----
#pragma unroll
        for (int t = 0; t < 4; ++t) {
            v16h vb = load_b_frag(vp, Sq, t * 16, kc, lane);
            o[t] = wmma16(pa.v, vb, o[t]);
        }
    }

    // ---- normalize + store ctx in [B,S,D] f16 ----
    const int b = bh >> 4;
    const int h = bh & 15;
#pragma unroll
    for (int r = 0; r < 8; ++r) {
        const float inv = 1.0f / lrow[r];
        const int srow = q0 + r + half * 8;
#pragma unroll
        for (int t = 0; t < 4; ++t) {
            const int d = h * HDm + t * 16 + mIdx;
            ctx[((size_t)b * Sq + srow) * Dm + d] = (_Float16)(o[t][r] * inv);
        }
    }
}

// ---------------------------------------------------------------------------
// Host-side launch
// ---------------------------------------------------------------------------
extern "C" void kernel_launch(void* const* d_in, const int* in_sizes, int n_in,
                              void* d_out, int out_size, void* d_ws, size_t ws_size,
                              hipStream_t stream) {
    const float* x  = (const float*)d_in[0];
    const float* Wq = (const float*)d_in[1];
    const float* bq = (const float*)d_in[2];
    const float* Wk = (const float*)d_in[3];
    const float* bk = (const float*)d_in[4];
    const float* Wv = (const float*)d_in[5];
    const float* bv = (const float*)d_in[6];
    const float* Wo = (const float*)d_in[7];
    const float* bo = (const float*)d_in[8];
    float* out = (float*)d_out;

    // workspace carve-up (f16 elements)
    _Float16* ws = (_Float16*)d_ws;
    size_t off = 0;
    _Float16* xh   = ws + off; off += (size_t)Mrows * Dm;       // 8192x1024
    _Float16* wqh  = ws + off; off += (size_t)Dm * Dm;
    _Float16* wkh  = ws + off; off += (size_t)Dm * Dm;
    _Float16* wvh  = ws + off; off += (size_t)Dm * Dm;
    _Float16* woh  = ws + off; off += (size_t)Dm * Dm;
    _Float16* qh   = ws + off; off += (size_t)Mrows * Dm;       // [B,H,S,HD]
    _Float16* kh   = ws + off; off += (size_t)Mrows * Dm;       // [B,H,S,HD]
    _Float16* vth  = ws + off; off += (size_t)Mrows * Dm;       // [B,H,HD,S]
    _Float16* ctxh = ws + off; off += (size_t)Mrows * Dm;       // [B,S,D]

    const int nX = Mrows * Dm;     // 8388608
    const int nW = Dm * Dm;        // 1048576

    cvt_f32_f16<<<(nX + 255) / 256, 256, 0, stream>>>(x,  xh,  nX);
    cvt_f32_f16<<<(nW + 255) / 256, 256, 0, stream>>>(Wq, wqh, nW);
    cvt_f32_f16<<<(nW + 255) / 256, 256, 0, stream>>>(Wk, wkh, nW);
    cvt_f32_f16<<<(nW + 255) / 256, 256, 0, stream>>>(Wv, wvh, nW);
    cvt_f32_f16<<<(nW + 255) / 256, 256, 0, stream>>>(Wo, woh, nW);

    dim3 ggrid(Mrows / 128, Dm / 128);   // 64 x 8
    proj_gemm<<<ggrid, 128, 0, stream>>>(xh, wqh, bq, qh,  nullptr, 0);
    proj_gemm<<<ggrid, 128, 0, stream>>>(xh, wkh, bk, kh,  nullptr, 0);
    proj_gemm<<<ggrid, 128, 0, stream>>>(xh, wvh, bv, vth, nullptr, 1);

    dim3 agrid(Sq / 128, Bsz * Hn);      // 16 x 64
    attn_kernel<<<agrid, 256, 0, stream>>>(qh, kh, vth, ctxh);

    proj_gemm<<<ggrid, 128, 0, stream>>>(ctxh, woh, bo, nullptr, out, 2);
}
